// MoCo_69114613727687
// MI455X (gfx1250) — matrix operand order repaired and checked by reading, not compiled, not run on hardware
//
#include <hip/hip_runtime.h>

typedef __attribute__((ext_vector_type(16))) _Float16 v16h;
typedef __attribute__((ext_vector_type(8)))  float    v8f;
typedef __attribute__((ext_vector_type(4)))  int      v4i;

#define B_TOK    16384
#define DDIM     4096
#define NEXP     4
#define ODIM     256
#define NTOT     1024      // NEXP*ODIM
#define BM       64        // rows per workgroup
#define KC       32        // K chunk (one WMMA k-step)
#define THREADS  512       // 16 waves (wave32)
#define BS_STRIDE  40      // halves per n-row in LDS (padded: 80B, 16B aligned, conflict-free)
#define ASH_STRIDE 40      // halves per A row in LDS
#define A32_STRIDE 33      // fp32 A row stride (bank-conflict pad)

union HalfPack { _Float16 h[2]; unsigned u; };
union Frag16   { v16h v; uint4 q[2]; };

// ---- CDNA5 async global->LDS path (ASYNCcnt), with safe fallback ----------
#if defined(__gfx1250__) && __has_builtin(__builtin_amdgcn_global_load_async_to_lds_b128)
#define HAVE_ASYNC_LDS 1
// builtin signature (from diagnostics): (int4 AS1*, int4 AS3*, imm, imm)
#define TO_GLB(p) ((__attribute__((address_space(1))) v4i*)(p))
#define TO_LDS(p) ((__attribute__((address_space(3))) v4i*)(p))
#else
#define HAVE_ASYNC_LDS 0
#endif

__device__ __forceinline__ void wait_asynccnt0() {
#if __has_builtin(__builtin_amdgcn_s_wait_asynccnt)
  __builtin_amdgcn_s_wait_asynccnt(0);
#else
  asm volatile("s_wait_asynccnt 0x0" ::: "memory");
#endif
}

// ---------------------------------------------------------------------------
// Pack expert_w [E,D,O] fp32 -> Bct [n = e*256+o][k] f16 (k-contiguous rows),
// LDS-tiled transpose so reads and writes are both vectorized.
// Grid: 1024 blocks x 256 threads; tile = 32 k x 128 n.
// ---------------------------------------------------------------------------
__global__ __launch_bounds__(256)
void pack_expert_weights(const float* __restrict__ ew, _Float16* __restrict__ bct) {
  __shared__ _Float16 tr[128 * 40];              // [n_local][k_local], padded stride
  const int bid = blockIdx.x;
  const int k0  = (bid & 127) * 32;
  const int n0  = (bid >> 7) * 128;              // stays within one expert (128 | 256)
  const int t   = threadIdx.x;
  #pragma unroll
  for (int i = 0; i < 16; ++i) {
    int idx = i * 256 + t;                       // 0..4095
    int kl  = idx >> 7;                          // 0..31
    int nl  = idx & 127;                         // 0..127  (coalesced over o)
    int n   = n0 + nl;
    int e   = n >> 8;
    int o   = n & 255;
    float v = ew[(size_t)e * DDIM * ODIM + (size_t)(k0 + kl) * ODIM + o];
    tr[nl * 40 + kl] = (_Float16)v;
  }
  __syncthreads();
  {
    int nl   = t >> 1;                           // 0..127
    int part = t & 1;                            // 32B halves each
    const uint4* src = (const uint4*)(tr + nl * 40 + part * 16);
    uint4* dst = (uint4*)(bct + (size_t)(n0 + nl) * DDIM + k0 + part * 16);
    dst[0] = src[0];
    dst[1] = src[1];
  }
}

// ---------------------------------------------------------------------------
// Fused MoE: per workgroup computes 64 rows x all 4 experts (A read once),
// router logits in fp32 VALU alongside WMMA, gating + combine in epilogue.
// ---------------------------------------------------------------------------
__global__ __launch_bounds__(THREADS)
void moe_fused(const float* __restrict__ A, const float* __restrict__ Rw,
               const _Float16* __restrict__ Bct, float* __restrict__ Out,
               float* __restrict__ RWout) {
  extern __shared__ char smem[];
  _Float16* BsT  = (_Float16*)smem;                    // 1024*40*2 = 81920 B
  _Float16* Ash  = (_Float16*)(smem + 81920);          // 64*40*2   =  5120 B
  float*    As32 = (float*)(smem + 87040);             // 64*33*4   =  8448 B
  float*    Rs   = (float*)(smem + 95488);             // 32*4*4    =   512 B
  float*    lg   = (float*)(smem + 96000);             // 64*4*4    =  1024 B
  float*    gsm  = (float*)(smem + 97024);             // 64*4*4    =  1024 B
  float*    Cred = (float*)smem;                       // aliases BsT (64*256*4 = 64 KB)

  const int t    = threadIdx.x;
  const int lane = t & 31;
  const int w    = t >> 5;        // 0..15
  const int mt   = w >> 2;        // m-tile 0..3
  const int eW   = w & 3;         // expert owned by this wave
  const int r0   = blockIdx.x * BM;

  v8f zero = {};
  v8f acc[16];
  #pragma unroll
  for (int j = 0; j < 16; ++j) acc[j] = zero;

  float lacc = 0.0f;              // router logit accumulator (threads < 256)

  const int arow = t >> 3;        // 0..63 : A loader row
  const int aq   = t & 7;         // 4-float segment
  // ISA 7.12.2 lane maps
  const int aM   = mt * 16 + (lane & 15);
  const int akb  = (lane >> 4) * 8;     // A: K halves {0..7,16..23} / {8..15,24..31}
  const int bkb  = (lane >> 4) * 8;     // B: K halves {0..7,16..23} / {8..15,24..31}
  const int bcol = lane & 15;

  for (int kc = 0; kc < DDIM / KC; ++kc) {
    const int k0 = kc * KC;
    __syncthreads();                              // prev compute done, LDS reusable
    // ---- B tile: Bct (L2-resident) -> BsT[n][0..31], async DMA to LDS ----
    #pragma unroll
    for (int i = 0; i < 8; ++i) {
      int o = i * THREADS + t;            // 0..4095 x b128
      int n = o >> 2;
      int part = o & 3;
      const _Float16* src = Bct + (size_t)n * DDIM + k0 + part * 8;
      _Float16* dst = BsT + n * BS_STRIDE + part * 8;
#if HAVE_ASYNC_LDS
      __builtin_amdgcn_global_load_async_to_lds_b128(TO_GLB(src), TO_LDS(dst), 0, 0);
#else
      *(uint4*)dst = *(const uint4*)src;
#endif
    }
    // ---- A tile: fp32 global -> LDS fp32 (for router) + f16 (for WMMA) ----
    {
      const float4 av = *(const float4*)(A + (size_t)(r0 + arow) * DDIM + k0 + aq * 4);
      float* a32 = As32 + arow * A32_STRIDE + aq * 4;
      a32[0] = av.x; a32[1] = av.y; a32[2] = av.z; a32[3] = av.w;
      HalfPack p0, p1;
      p0.h[0] = (_Float16)av.x; p0.h[1] = (_Float16)av.y;
      p1.h[0] = (_Float16)av.z; p1.h[1] = (_Float16)av.w;
      unsigned* ah = (unsigned*)(Ash + arow * ASH_STRIDE + aq * 4);
      ah[0] = p0.u; ah[1] = p1.u;
    }
    // ---- router chunk (contiguous 128 fp32) ----
    if (t < KC * NEXP) Rs[t] = Rw[(size_t)k0 * NEXP + t];
#if HAVE_ASYNC_LDS
    wait_asynccnt0();                             // our async stores to LDS visible
#endif
    __syncthreads();
    // ---- fp32 router logits (VALU; co-executes with matrix pipe) ----
    if (t < BM * NEXP) {
      const int rr = t >> 2, ee = t & 3;
      float s = lacc;
      #pragma unroll
      for (int k = 0; k < KC; ++k)
        s = fmaf(As32[rr * A32_STRIDE + k], Rs[k * NEXP + ee], s);
      lacc = s;
    }
    // ---- one WMMA k-step: 1 A frag reused across 16 n-tiles ----
    Frag16 af;
    af.q[0] = *(const uint4*)(Ash + aM * ASH_STRIDE + akb);
    af.q[1] = *(const uint4*)(Ash + aM * ASH_STRIDE + akb + 16);
    #pragma unroll
    for (int j = 0; j < 16; ++j) {
      Frag16 bf;
      const int n = eW * ODIM + j * 16 + bcol;
      const uint4* bp = (const uint4*)(BsT + n * BS_STRIDE + bkb);
      bf.q[0] = bp[0];
      bf.q[1] = bp[2];
      acc[j] = __builtin_amdgcn_wmma_f32_16x16x32_f16(
          false, af.v, false, bf.v, (short)0, acc[j], false, false);
    }
  }

  // ---- gating: softmax (fp32) + top-2 + renorm ----
  __syncthreads();
  if (t < BM * NEXP) lg[t] = lacc;
  __syncthreads();
  if (t < BM) {
    float l0 = lg[t*4+0], l1 = lg[t*4+1], l2 = lg[t*4+2], l3 = lg[t*4+3];
    float m  = fmaxf(fmaxf(l0, l1), fmaxf(l2, l3));
    float rw[4];
    rw[0] = expf(l0 - m); rw[1] = expf(l1 - m);
    rw[2] = expf(l2 - m); rw[3] = expf(l3 - m);
    float inv = 1.0f / (rw[0] + rw[1] + rw[2] + rw[3]);
    rw[0] *= inv; rw[1] *= inv; rw[2] *= inv; rw[3] *= inv;
    float4 rwv; rwv.x = rw[0]; rwv.y = rw[1]; rwv.z = rw[2]; rwv.w = rw[3];
    *(float4*)(RWout + (size_t)(r0 + t) * NEXP) = rwv;   // routing_weights output
    int i0 = 0;                                          // strict > : first index wins ties
    if (rw[1] > rw[i0]) i0 = 1;
    if (rw[2] > rw[i0]) i0 = 2;
    if (rw[3] > rw[i0]) i0 = 3;
    int i1 = -1; float b = -1.0f;
    #pragma unroll
    for (int e2 = 0; e2 < 4; ++e2) {
      if (e2 == i0) continue;
      if (rw[e2] > b) { b = rw[e2]; i1 = e2; }
    }
    float s2 = 1.0f / (rw[i0] + rw[i1]);
    float g[4] = {0.0f, 0.0f, 0.0f, 0.0f};
    g[i0] = rw[i0] * s2;
    g[i1] = rw[i1] * s2;
    gsm[t*4+0] = g[0]; gsm[t*4+1] = g[1]; gsm[t*4+2] = g[2]; gsm[t*4+3] = g[3];
  }
  __syncthreads();
  // ---- zero the reduction buffer (aliases dead B tile) ----
  #pragma unroll
  for (int i = 0; i < 32; ++i) Cred[i * THREADS + t] = 0.0f;
  __syncthreads();
  // ---- gate-scaled cross-expert combine via ds_add_f32 ----
  {
    const int rofs = (lane >> 4) * 8;          // C/D layout: lanes 16-31 hold M+8
    float gv[8];
    #pragma unroll
    for (int v = 0; v < 8; ++v) gv[v] = gsm[(mt * 16 + v + rofs) * 4 + eW];
    #pragma unroll
    for (int j = 0; j < 16; ++j) {
      #pragma unroll
      for (int v = 0; v < 8; ++v) {
        int row = mt * 16 + v + rofs;
        atomicAdd(&Cred[row * ODIM + j * 16 + bcol], acc[j][v] * gv[v]);
      }
    }
  }
  __syncthreads();
  // ---- coalesced float4 writeout of moe_output ----
  #pragma unroll
  for (int i = 0; i < 8; ++i) {
    int idx = i * THREADS + t;                 // 0..4095 float4 units
    int row = idx >> 6;
    int c4  = idx & 63;
    *(float4*)(Out + (size_t)(r0 + row) * ODIM + c4 * 4) =
        *(const float4*)(Cred + row * ODIM + c4 * 4);
  }
}

// ---------------------------------------------------------------------------
extern "C" void kernel_launch(void* const* d_in, const int* in_sizes, int n_in,
                              void* d_out, int out_size, void* d_ws, size_t ws_size,
                              hipStream_t stream) {
  (void)in_sizes; (void)n_in; (void)out_size; (void)ws_size;
  const float* A  = (const float*)d_in[0];    // v_standard [16384,4096]
  const float* Rw = (const float*)d_in[1];    // router_w   [4096,4]
  const float* Ew = (const float*)d_in[2];    // expert_w   [4,4096,256]
  float* Out   = (float*)d_out;                         // moe_output [16384,256]
  float* RWout = Out + (size_t)B_TOK * ODIM;            // routing_weights [16384,4]
  _Float16* Bct = (_Float16*)d_ws;                      // 8 MiB packed f16 weights

  pack_expert_weights<<<1024, 256, 0, stream>>>(Ew, Bct);

  const size_t shmem = 98048;                           // ~96 KB dynamic LDS
  moe_fused<<<B_TOK / BM, THREADS, shmem, stream>>>(A, Rw, Bct, Out, RWout);
}
